// DeformTransformer_506806141302
// MI455X (gfx1250) — matrix operand order repaired and checked by reading, not compiled, not run on hardware
//
#include <hip/hip_runtime.h>
#include <hip/hip_bf16.h>

// ---------------------------------------------------------------------------
// Deformable-DETR transformer forward for MI455X (gfx1250, wave32, WMMA).
// Dense GEMMs: fp32 -> bf16 staged via LDS, v_wmma_f32_16x16x32_bf16, f32 acc.
// Loads are unconditional global_load_b128 (bounds by address clamping, legal
// because OOB A-rows / W-cols only feed never-stored C entries), with a
// register-staged pipeline so the next K-tile's loads overlap current WMMAs.
// ---------------------------------------------------------------------------

typedef __attribute__((ext_vector_type(16))) __bf16 v16bf;
typedef __attribute__((ext_vector_type(8)))  __bf16 v8bf;
typedef __attribute__((ext_vector_type(4)))  __bf16 v4bf;
typedef __attribute__((ext_vector_type(8)))  float  v8f;

#define NHEADS 8
#define HEADD  32
#define NPTS   4

struct Lvls { int h[4]; int w[4]; int start[4]; };

// ---------------------------------------------------------------------------
// GEMM: C[M,N] = act( A(M,K) * W(K,N) + bias[N] )
// TRANS==0: A row-major, element (m,k) = A[m*lda + k]
// TRANS==1: A column-contiguous, element (m,k) = A[k*lda + m]  (input proj)
// W row-major fp32 [K,N]. Requires K%32==0 and M,N multiples of 4.
// Block: 64x64 tile, 128 threads (4 waves); wave = 32x32 = 2x2 WMMA tiles.
// ---------------------------------------------------------------------------
template<int ACT, int TRANS>
__global__ __launch_bounds__(128)
void gemm_bf16_wmma(const float* __restrict__ A, long lda,
                    const float* __restrict__ W, const float* __restrict__ bias,
                    float* __restrict__ C, int M, int N, int K)
{
  __shared__ __align__(16) __bf16 As[64][40];   // 32 cols + pad (16B rows)
  __shared__ __align__(16) __bf16 Bs[32][80];   // 64 cols + pad (16B rows)

  const int t    = threadIdx.x;
  const int lane = t & 31;
  const int wid  = t >> 5;
  const int wm   = wid >> 1;
  const int wn   = wid & 1;
  const int bm   = blockIdx.y * 64;
  const int bn   = blockIdx.x * 64;

  v8f acc[2][2];
  const v8f vzero = {0.f,0.f,0.f,0.f,0.f,0.f,0.f,0.f};
  acc[0][0] = vzero; acc[0][1] = vzero; acc[1][0] = vzero; acc[1][1] = vzero;

  // ---- staging coordinates (computed once; unconditional b128 loads) ----
  const float* aPtr[4]; const float* wPtr[4];
  int aRow[4], aCol[4], wRow[4], wCol[4];
#pragma unroll
  for (int i = 0; i < 4; ++i) {
    int idx = i * 128 + t;
    if (!TRANS) {
      // A tile 64x32: chunk = 4 consecutive k of one row
      int r = idx >> 3, c4 = (idx & 7) * 4;
      int m = bm + r; if (m >= M) m = M - 1;            // clamp: row unused
      aRow[i] = r; aCol[i] = c4;
      aPtr[i] = A + (long)m * lda + c4;
    } else {
      // A tile 64x32: chunk = 4 consecutive m of one k column (contiguous)
      int c = idx >> 4, m4 = (idx & 15) * 4;
      int m = bm + m4; if (m + 3 >= M) m = 0;           // redirect OOB chunk
      aRow[i] = m4; aCol[i] = c;
      aPtr[i] = A + (long)c * lda + m;
    }
    // W tile 32x64: chunk = 4 consecutive n of one k row
    int r = idx >> 4, c4 = (idx & 15) * 4;
    int n = bn + c4; if (n + 3 >= N) n = 0;             // redirect OOB chunk
    wRow[i] = r; wCol[i] = c4;
    wPtr[i] = W + (long)r * N + n;
  }

  float4 ra[4], rw[4];
#pragma unroll
  for (int i = 0; i < 4; ++i) { ra[i] = *(const float4*)aPtr[i]; rw[i] = *(const float4*)wPtr[i]; }

  for (int k0 = 0; k0 < K; k0 += 32) {
    // commit staged registers to LDS (fp32 -> bf16)
#pragma unroll
    for (int i = 0; i < 4; ++i) {
      if (!TRANS) {
        v4bf h; h[0] = (__bf16)ra[i].x; h[1] = (__bf16)ra[i].y;
                h[2] = (__bf16)ra[i].z; h[3] = (__bf16)ra[i].w;
        *(v4bf*)&As[aRow[i]][aCol[i]] = h;
      } else {
        As[aRow[i] + 0][aCol[i]] = (__bf16)ra[i].x;
        As[aRow[i] + 1][aCol[i]] = (__bf16)ra[i].y;
        As[aRow[i] + 2][aCol[i]] = (__bf16)ra[i].z;
        As[aRow[i] + 3][aCol[i]] = (__bf16)ra[i].w;
      }
      v4bf g; g[0] = (__bf16)rw[i].x; g[1] = (__bf16)rw[i].y;
              g[2] = (__bf16)rw[i].z; g[3] = (__bf16)rw[i].w;
      *(v4bf*)&Bs[wRow[i]][wCol[i]] = g;
    }
    __syncthreads();

    // issue next tile's global loads early; they overlap ds_load + wmma below
    if (k0 + 32 < K) {
#pragma unroll
      for (int i = 0; i < 4; ++i) {
        aPtr[i] += TRANS ? 32 * lda : 32;
        wPtr[i] += (long)32 * N;
        ra[i] = *(const float4*)aPtr[i];
        rw[i] = *(const float4*)wPtr[i];
      }
    }

    // A fragments: lane L -> M = L%16; lanes 0-15 hold K[0..7]+[16..23],
    // lanes 16-31 hold K[8..15]+[24..31] (ISA 7.12.2, 16-bit A 16x32).
    const int half8 = (lane >> 4) * 8;
    v16bf afr[2], bfr[2];
#pragma unroll
    for (int ti = 0; ti < 2; ++ti) {
      int m = wm * 32 + ti * 16 + (lane & 15);
      v8bf lo = *(const v8bf*)&As[m][half8];
      v8bf hi = *(const v8bf*)&As[m][16 + half8];
      afr[ti] = __builtin_shufflevector(lo, hi, 0,1,2,3,4,5,6,7,8,9,10,11,12,13,14,15);
    }
    // B fragments: lane L -> N = L%16; elements hold K in the same order as A.
#pragma unroll
    for (int tj = 0; tj < 2; ++tj) {
      int n = wn * 32 + tj * 16 + (lane & 15);
#pragma unroll
      for (int s = 0; s < 8; ++s) {
        bfr[tj][s]     = Bs[half8 + s][n];
        bfr[tj][8 + s] = Bs[16 + half8 + s][n];
      }
    }
#pragma unroll
    for (int ti = 0; ti < 2; ++ti)
#pragma unroll
      for (int tj = 0; tj < 2; ++tj)
        acc[ti][tj] = __builtin_amdgcn_wmma_f32_16x16x32_bf16(
            false, afr[ti], false, bfr[tj], (short)0, acc[ti][tj], false, false);
    __syncthreads();
  }

  // epilogue: C/D layout -> N = lane%16, M = vgpr + 8*(lane/16); half-wave
  // stores are 16 consecutive n -> coalesced 64B runs.
  const int half = lane >> 4;
#pragma unroll
  for (int ti = 0; ti < 2; ++ti)
#pragma unroll
    for (int tj = 0; tj < 2; ++tj) {
      int n = bn + wn * 32 + tj * 16 + (lane & 15);
      if (n < N) {
        float bz = bias[n];
#pragma unroll
        for (int v = 0; v < 8; ++v) {
          int m = bm + wm * 32 + ti * 16 + v + 8 * half;
          if (m < M) {
            float o = acc[ti][tj][v] + bz;
            if (ACT) o = fmaxf(o, 0.f);
            C[(long)m * N + n] = o;
          }
        }
      }
    }
}

// ---------------------------------------------------------------------------
// Fused residual-add + LayerNorm over D=256; one wave32 per row.
// ---------------------------------------------------------------------------
__global__ __launch_bounds__(128)
void add_layernorm(const float* __restrict__ x, const float* __restrict__ r,
                   const float* __restrict__ g, const float* __restrict__ b,
                   float* __restrict__ o, int M)
{
  int row = (blockIdx.x * blockDim.x + threadIdx.x) >> 5;
  int lane = threadIdx.x & 31;
  if (row >= M) return;
  const float* xr = x + (long)row * 256;
  const float* rr = r + (long)row * 256;
  float v[8]; float s = 0.f, sq = 0.f;
#pragma unroll
  for (int j = 0; j < 8; ++j) {
    float t = xr[lane + 32 * j] + rr[lane + 32 * j];
    v[j] = t; s += t; sq += t * t;
  }
#pragma unroll
  for (int off = 16; off > 0; off >>= 1) {
    s  += __shfl_xor(s,  off, 32);
    sq += __shfl_xor(sq, off, 32);
  }
  float mean = s * (1.f / 256.f);
  float var  = sq * (1.f / 256.f) - mean * mean;
  float rstd = rsqrtf(var + 1e-5f);
  float* orow = o + (long)row * 256;
#pragma unroll
  for (int j = 0; j < 8; ++j) {
    int c = lane + 32 * j;
    orow[c] = (v[j] - mean) * rstd * g[c] + b[c];
  }
}

// ---------------------------------------------------------------------------
// Multi-scale deformable attention sampling.
// One wave32 per (b, query, head); lane = head channel (HD == 32 == wave).
// ---------------------------------------------------------------------------
__global__ __launch_bounds__(128)
void msda_sample(const float* __restrict__ value,              // [Bn,Sv,256]
                 const float* __restrict__ off, int offStride,  // [Bn,Lq,HH*nl*P*2]
                 const float* __restrict__ attL, int attStride, // [Bn,Lq,HH*nl*P]
                 const float* __restrict__ ref, int refBatched, // [Lq,2] or [Bn,Lq,2]
                 Lvls lv, int nl, int Bn, int Lq, int Sv,
                 float* __restrict__ out)                       // [Bn,Lq,256]
{
  int gw   = (blockIdx.x * blockDim.x + threadIdx.x) >> 5;
  int lane = threadIdx.x & 31;
  int total = Bn * Lq * NHEADS;
  if (gw >= total) return;
  int b   = gw / (Lq * NHEADS);
  int rem = gw - b * Lq * NHEADS;
  int q   = rem >> 3;
  int h   = rem & 7;
  int np  = nl * NPTS;

  const float* orow = off  + (long)(b * Lq + q) * offStride + h * np * 2;
  const float* arow = attL + (long)(b * Lq + q) * attStride + h * np;

  float wts[16];
  float mx = -1e30f;
  for (int j = 0; j < np; ++j) { wts[j] = arow[j]; mx = fmaxf(mx, wts[j]); }
  float ssum = 0.f;
  for (int j = 0; j < np; ++j) { wts[j] = __expf(wts[j] - mx); ssum += wts[j]; }
  float inv = 1.f / ssum;

  const float* rp = refBatched ? (ref + (long)(b * Lq + q) * 2) : (ref + (long)q * 2);
  float rx = rp[0], ry = rp[1];

  const float* vb = value + (long)b * Sv * 256 + h * HEADD + lane;
  float acc = 0.f;

  for (int l = 0; l < nl; ++l) {
    const int   W_ = lv.w[l], H_ = lv.h[l], st = lv.start[l];
    const float fw = (float)W_, fh = (float)H_;
    for (int p = 0; p < NPTS; ++p) {
      int j = l * NPTS + p;
      float wt   = wts[j] * inv;
      float locx = rx + orow[j * 2]     / fw;
      float locy = ry + orow[j * 2 + 1] / fh;
      float x = locx * fw - 0.5f, y = locy * fh - 0.5f;
      float x0f = floorf(x), y0f = floorf(y);
      float lx = x - x0f, ly = y - y0f;
      int x0 = (int)x0f, y0 = (int)y0f;
      float w00 = (1.f - lx) * (1.f - ly) * wt;
      float w10 = lx * (1.f - ly) * wt;
      float w01 = (1.f - lx) * ly * wt;
      float w11 = lx * ly * wt;
      if (x0 >= 0 && x0 < W_ && y0 >= 0 && y0 < H_)
        acc += w00 * vb[(long)(st + y0 * W_ + x0) * 256];
      if (x0 + 1 >= 0 && x0 + 1 < W_ && y0 >= 0 && y0 < H_)
        acc += w10 * vb[(long)(st + y0 * W_ + x0 + 1) * 256];
      if (x0 >= 0 && x0 < W_ && y0 + 1 >= 0 && y0 + 1 < H_)
        acc += w01 * vb[(long)(st + (y0 + 1) * W_ + x0) * 256];
      if (x0 + 1 >= 0 && x0 + 1 < W_ && y0 + 1 >= 0 && y0 + 1 < H_)
        acc += w11 * vb[(long)(st + (y0 + 1) * W_ + x0 + 1) * 256];
    }
  }
  out[(long)(b * Lq + q) * 256 + h * HEADD + lane] = acc;
}

// normalized grid-center reference points over concatenated levels
__global__ void grid_ref_kernel(float* __restrict__ out, Lvls lv, int nl, int total)
{
  int s = blockIdx.x * blockDim.x + threadIdx.x;
  if (s >= total) return;
  int l = 0, off = s;
  while (l < nl - 1 && off >= lv.h[l] * lv.w[l]) { off -= lv.h[l] * lv.w[l]; ++l; }
  int W_ = lv.w[l];
  int iy = off / W_, ix = off - iy * W_;
  out[s * 2]     = (ix + 0.5f) / (float)W_;
  out[s * 2 + 1] = (iy + 0.5f) / (float)lv.h[l];
}

// o[i] = a[i] + b[i mod bmod], specialized for n == 2*bmod (batch of 2)
__global__ void add_bcast_kernel(const float* __restrict__ a, const float* __restrict__ b,
                                 float* __restrict__ o, long n, long bmod)
{
  long i = (long)blockIdx.x * blockDim.x + threadIdx.x;
  if (i < n) { long j = (i >= bmod) ? i - bmod : i; o[i] = a[i] + b[j]; }
}

__global__ void bcast_copy_kernel(const float* __restrict__ a, float* __restrict__ o,
                                  long n, long amod)
{
  long i = (long)blockIdx.x * blockDim.x + threadIdx.x;
  if (i < n) { long j = (i >= amod) ? i - amod : i; o[i] = a[j]; }
}

// ---------------------------------------------------------------------------
// Host orchestration
// ---------------------------------------------------------------------------
static inline void launch_gemm(hipStream_t s, const float* A, long lda, bool trans,
                               const float* W, const float* bias, float* C,
                               int M, int N, int K, bool relu)
{
  dim3 g((N + 63) / 64, (M + 63) / 64);
  if (trans) {
    if (relu) gemm_bf16_wmma<1,1><<<g, 128, 0, s>>>(A, lda, W, bias, C, M, N, K);
    else      gemm_bf16_wmma<0,1><<<g, 128, 0, s>>>(A, lda, W, bias, C, M, N, K);
  } else {
    if (relu) gemm_bf16_wmma<1,0><<<g, 128, 0, s>>>(A, lda, W, bias, C, M, N, K);
    else      gemm_bf16_wmma<0,0><<<g, 128, 0, s>>>(A, lda, W, bias, C, M, N, K);
  }
}

extern "C" void kernel_launch(void* const* d_in, const int* in_sizes, int n_in,
                              void* d_out, int out_size, void* d_ws, size_t ws_size,
                              hipStream_t stream)
{
  (void)in_sizes; (void)n_in; (void)out_size; (void)ws_size;
  const int Bn = 2, Sn = 12240, Tn = 1024;

  int idx = 0;
  auto nxt = [&]() { return (const float*)d_in[idx++]; };
  // setup_inputs() insertion order, nested dicts DFS in insertion order
  const float* srcs[4] = { nxt(), nxt(), nxt(), nxt() };
  const float* t2s_ref = nxt();
  const float* proj_w  = nxt();
  const float* proj_b  = nxt();
  // encoder params
  const float *e_vw = nxt(), *e_vb = nxt(), *e_ow = nxt(), *e_ob = nxt();
  const float *e_aw = nxt(), *e_ab = nxt(), *e_uw = nxt(), *e_ub = nxt();
  const float *e_n1g = nxt(), *e_n1b = nxt();
  const float *e_l1w = nxt(), *e_l1b = nxt(), *e_l2w = nxt(), *e_l2b = nxt();
  const float *e_n2g = nxt(), *e_n2b = nxt();
  // decoder params
  const float *dsa_vw = nxt(), *dsa_vb = nxt(), *dsa_ow = nxt(), *dsa_ob = nxt();
  const float *dsa_aw = nxt(), *dsa_ab = nxt(), *dsa_uw = nxt(), *dsa_ub = nxt();
  const float *dca_vw = nxt(), *dca_vb = nxt(), *dca_ow = nxt(), *dca_ob = nxt();
  const float *dca_aw = nxt(), *dca_ab = nxt(), *dca_uw = nxt(), *dca_ub = nxt();
  const float *d_n1g = nxt(), *d_n1b = nxt(), *d_n2g = nxt(), *d_n2b = nxt();
  const float *d_l1w = nxt(), *d_l1b = nxt(), *d_l2w = nxt(), *d_l2b = nxt();
  const float *d_n3g = nxt(), *d_n3b = nxt();
  const float *tgt_embed = nxt(), *tgt_pos = nxt(), *src_pos = nxt();

  // workspace layout
  float* ws = (float*)d_ws;
  size_t cur = 0;
  auto alloc = [&](size_t n) { float* p = ws + cur; cur += (n + 63) & ~(size_t)63; return p; };
  const size_t NSD = (size_t)Bn * Sn * 256;
  float* x   = alloc(NSD);                       // running encoder activations / memory
  float* qb  = alloc(NSD);                       // query (x+pos); reused as sampling output
  float* val = alloc(NSD);                       // value projection
  float* off = alloc(NSD);                       // sampling offsets
  float* att = alloc((size_t)Bn * Sn * 128);     // attention logits
  float* x2  = alloc(NSD);                       // attn/FFN branch output
  const int CH = 6144;                           // FFN row-chunk
  float* ffn = alloc((size_t)CH * 1024);
  float* yb  = alloc((size_t)Bn * Tn * 256);     // decoder activations
  float* eref = alloc((size_t)Sn * 2);
  float* tref = alloc((size_t)Tn * 2);
  float* smp = qb;                               // sampling output aliases query buffer

  Lvls lv4, lv1;
  const int hs[4] = {96, 48, 24, 12};
  const int starts[4] = {0, 9216, 11520, 12096};
  for (int l = 0; l < 4; ++l) { lv4.h[l] = hs[l]; lv4.w[l] = hs[l]; lv4.start[l] = starts[l]; }
  for (int l = 0; l < 4; ++l) { lv1.h[l] = 1; lv1.w[l] = 1; lv1.start[l] = 0; }
  lv1.h[0] = 32; lv1.w[0] = 32; lv1.start[0] = 0;

  grid_ref_kernel<<<(Sn + 255) / 256, 256, 0, stream>>>(eref, lv4, 4, Sn);
  grid_ref_kernel<<<(Tn + 255) / 256, 256, 0, stream>>>(tref, lv1, 1, Tn);

  // ---- input projection: x[b, start_l + m, :] = src_l[b,:,m]^T @ proj_w[l] + proj_b[l]
  for (int l = 0; l < 4; ++l) {
    int HW = hs[l] * hs[l];
    for (int b = 0; b < Bn; ++b)
      launch_gemm(stream, srcs[l] + (size_t)b * 256 * HW, /*lda=*/HW, /*trans=*/true,
                  proj_w + (size_t)l * 256 * 256, proj_b + (size_t)l * 256,
                  x + ((size_t)b * Sn + starts[l]) * 256, HW, 256, 256, false);
  }

  const long nEnc = (long)Bn * Sn * 256;
  const int lnBlocksE = (Bn * Sn + 3) / 4;
  const int sampBlocksE = (Bn * Sn * NHEADS + 3) / 4;

  // ---- encoder layers
  for (int li = 0; li < 3; ++li) {
    const float* vw = e_vw + (size_t)li * 256 * 256; const float* vb = e_vb + (size_t)li * 256;
    const float* ow = e_ow + (size_t)li * 256 * 256; const float* ob = e_ob + (size_t)li * 256;
    const float* aw = e_aw + (size_t)li * 256 * 128; const float* ab = e_ab + (size_t)li * 128;
    const float* uw = e_uw + (size_t)li * 256 * 256; const float* ub = e_ub + (size_t)li * 256;

    add_bcast_kernel<<<(nEnc + 255) / 256, 256, 0, stream>>>(x, src_pos, qb, nEnc, (long)Sn * 256);
    launch_gemm(stream, x,  256, false, vw, vb, val, Bn * Sn, 256, 256, false);
    launch_gemm(stream, qb, 256, false, ow, ob, off, Bn * Sn, 256, 256, false);
    launch_gemm(stream, qb, 256, false, aw, ab, att, Bn * Sn, 128, 256, false);
    msda_sample<<<sampBlocksE, 128, 0, stream>>>(val, off, 256, att, 128, eref, 0,
                                                 lv4, 4, Bn, Sn, Sn, smp);
    launch_gemm(stream, smp, 256, false, uw, ub, x2, Bn * Sn, 256, 256, false);
    add_layernorm<<<lnBlocksE, 128, 0, stream>>>(x, x2,
        e_n1g + (size_t)li * 256, e_n1b + (size_t)li * 256, x, Bn * Sn);
    for (int r0 = 0; r0 < Bn * Sn; r0 += CH) {
      int mrows = Bn * Sn - r0 < CH ? Bn * Sn - r0 : CH;
      launch_gemm(stream, x + (size_t)r0 * 256, 256, false,
                  e_l1w + (size_t)li * 256 * 1024, e_l1b + (size_t)li * 1024,
                  ffn, mrows, 1024, 256, true);
      launch_gemm(stream, ffn, 1024, false,
                  e_l2w + (size_t)li * 1024 * 256, e_l2b + (size_t)li * 256,
                  x2 + (size_t)r0 * 256, mrows, 256, 1024, false);
    }
    add_layernorm<<<lnBlocksE, 128, 0, stream>>>(x, x2,
        e_n2g + (size_t)li * 256, e_n2b + (size_t)li * 256, x, Bn * Sn);
  }
  // x is now `memory`

  // ---- decoder
  const long nDec = (long)Bn * Tn * 256;
  const int lnBlocksD = (Bn * Tn + 3) / 4;
  const int sampBlocksD = (Bn * Tn * NHEADS + 3) / 4;
  bcast_copy_kernel<<<(nDec + 255) / 256, 256, 0, stream>>>(tgt_embed, yb, nDec, (long)Tn * 256);

  for (int li = 0; li < 3; ++li) {
    // self-attention (single level 32x32 over the queries themselves)
    add_bcast_kernel<<<(nDec + 255) / 256, 256, 0, stream>>>(yb, tgt_pos, qb, nDec, (long)Tn * 256);
    launch_gemm(stream, yb, 256, false, dsa_vw + (size_t)li * 256 * 256, dsa_vb + (size_t)li * 256,
                val, Bn * Tn, 256, 256, false);
    launch_gemm(stream, qb, 256, false, dsa_ow + (size_t)li * 256 * 64, dsa_ob + (size_t)li * 64,
                off, Bn * Tn, 64, 256, false);
    launch_gemm(stream, qb, 256, false, dsa_aw + (size_t)li * 256 * 32, dsa_ab + (size_t)li * 32,
                att, Bn * Tn, 32, 256, false);
    msda_sample<<<sampBlocksD, 128, 0, stream>>>(val, off, 64, att, 32, tref, 0,
                                                 lv1, 1, Bn, Tn, Tn, smp);
    launch_gemm(stream, smp, 256, false, dsa_uw + (size_t)li * 256 * 256, dsa_ub + (size_t)li * 256,
                x2, Bn * Tn, 256, 256, false);
    add_layernorm<<<lnBlocksD, 128, 0, stream>>>(yb, x2,
        d_n2g + (size_t)li * 256, d_n2b + (size_t)li * 256, yb, Bn * Tn);

    // cross-attention into encoder memory (4 levels)
    add_bcast_kernel<<<(nDec + 255) / 256, 256, 0, stream>>>(yb, tgt_pos, qb, nDec, (long)Tn * 256);
    launch_gemm(stream, x, 256, false, dca_vw + (size_t)li * 256 * 256, dca_vb + (size_t)li * 256,
                val, Bn * Sn, 256, 256, false);
    launch_gemm(stream, qb, 256, false, dca_ow + (size_t)li * 256 * 256, dca_ob + (size_t)li * 256,
                off, Bn * Tn, 256, 256, false);
    launch_gemm(stream, qb, 256, false, dca_aw + (size_t)li * 256 * 128, dca_ab + (size_t)li * 128,
                att, Bn * Tn, 128, 256, false);
    msda_sample<<<sampBlocksD, 128, 0, stream>>>(val, off, 256, att, 128, t2s_ref, 1,
                                                 lv4, 4, Bn, Tn, Sn, smp);
    launch_gemm(stream, smp, 256, false, dca_uw + (size_t)li * 256 * 256, dca_ub + (size_t)li * 256,
                x2, Bn * Tn, 256, 256, false);
    add_layernorm<<<lnBlocksD, 128, 0, stream>>>(yb, x2,
        d_n1g + (size_t)li * 256, d_n1b + (size_t)li * 256, yb, Bn * Tn);

    // FFN
    launch_gemm(stream, yb, 256, false, d_l1w + (size_t)li * 256 * 1024, d_l1b + (size_t)li * 1024,
                ffn, Bn * Tn, 1024, 256, true);
    launch_gemm(stream, ffn, 1024, false, d_l2w + (size_t)li * 1024 * 256, d_l2b + (size_t)li * 256,
                x2, Bn * Tn, 256, 1024, false);
    float* dst = (li == 2) ? (float*)d_out : yb;   // final LN writes output directly
    add_layernorm<<<lnBlocksD, 128, 0, stream>>>(yb, x2,
        d_n3g + (size_t)li * 256, d_n3b + (size_t)li * 256, dst, Bn * Tn);
  }
}
